// _DVNCCodebook_44178033606669
// MI455X (gfx1250) — matrix operand order repaired and checked by reading, not compiled, not run on hardware
//
#include <hip/hip_runtime.h>
#include <stdint.h>

// ---- CDNA5 WMMA types ----
typedef __attribute__((ext_vector_type(16))) __bf16 v16bf;
typedef __attribute__((ext_vector_type(8)))  float  v8f;

#define DDIM 256
#define NCODES 512
#define ROWS_PER_BLOCK 128
#define NBLOCKS 512            // 65536 rows / 128
#define HZ_STRIDE 260          // 256 + 4 floats pad: kills LDS bank conflicts
#define LN_EPS 1e-5f
#define LOSS_SCALE (1.25f / (65536.0f * 256.0f))   // (1+BETA) * mean

static __device__ __forceinline__ v8f wmma_bf16(v16bf a, v16bf b, v8f c) {
    return __builtin_amdgcn_wmma_f32_16x16x32_bf16(false, a, false, b, (short)0, c,
                                                   false, false);
}

// ---------------------------------------------------------------------------
// K0: precompute bf16 copies of W_in / W_out / codebook, codebook norms |c|^2,
//     and zero the loss accumulator (required for deterministic graph replay).
// grid: 512 x 256 threads  (gid 0..131071)
// ---------------------------------------------------------------------------
__global__ __launch_bounds__(256) void vq_init_kernel(
    const float* __restrict__ codebook, const float* __restrict__ W_in,
    const float* __restrict__ W_out,
    __bf16* __restrict__ winb, __bf16* __restrict__ woutb,
    __bf16* __restrict__ cbb, float* __restrict__ cc,
    float* __restrict__ loss)
{
    const int gid = blockIdx.x * 256 + threadIdx.x;
    if (gid < 65536) {
        winb[gid]  = (__bf16)W_in[gid];
        woutb[gid] = (__bf16)W_out[gid];
    }
    if (gid < NCODES * DDIM) cbb[gid] = (__bf16)codebook[gid];
    if (gid < NCODES) {
        const float* r = codebook + (size_t)gid * DDIM;
        float s = 0.f;
        #pragma unroll 8
        for (int j = 0; j < DDIM; ++j) { float v = r[j]; s += v * v; }
        cc[gid] = s;
    }
    if (gid == 0) *loss = 0.f;
}

// ---------------------------------------------------------------------------
// K1: fused  z = H @ Win^T -> argmin dist to codes -> loss -> zq @ Wout^T
//           -> masked residual -> LayerNorm.
// 512 blocks x 256 threads (8 waves). Each wave owns a 16-row tile.
// ---------------------------------------------------------------------------
__global__ __launch_bounds__(256) void vq_fused_kernel(
    const float*  __restrict__ hidden,
    const uint8_t* __restrict__ amask,       // bool array, 1 byte/elem
    const float*  __restrict__ codebook,     // f32, for exact loss
    const float*  __restrict__ gamma,
    const float*  __restrict__ beta,
    const __bf16* __restrict__ winb,
    const __bf16* __restrict__ woutb,
    const __bf16* __restrict__ cbb,
    const float*  __restrict__ cc,
    float* __restrict__ out,
    float* __restrict__ loss)
{
    __shared__ float hz[8][16][HZ_STRIDE];   // per-wave z tile, then h tile
    __shared__ float ccs[NCODES];
    __shared__ float gms[DDIM], bts[DDIM];
    __shared__ float zzs[8][16];             // per-row |z|^2
    __shared__ int   idxs[8][16];            // per-row argmin code

    const int t    = threadIdx.x;
    const int w    = t >> 5;                 // wave id 0..7
    const int l    = t & 31;                 // lane
    const int half = l >> 4;                 // 0: lanes 0-15, 1: lanes 16-31
    const int lc   = l & 15;
    const int rowBase = blockIdx.x * ROWS_PER_BLOCK + w * 16;

    gms[t] = gamma[t];  bts[t] = beta[t];
    ccs[t] = cc[t];     ccs[t + 256] = cc[t + 256];
    __syncthreads();

    // ================= Stage A: z = H @ Win^T ==========================
    // A-fragment k order used here (contiguous; B matches): lanes 0-15 row=lc hold
    // K 0..15; lanes 16-31 row=lc hold K 16..31.
    v16bf afrag[8];
    {
        const float* hrow = hidden + (size_t)(rowBase + lc) * DDIM;
        #pragma unroll
        for (int k = 0; k < 8; ++k) {
            const float* p0 = hrow + k * 32 + half * 16;
            v16bf a;
            #pragma unroll
            for (int j = 0; j < 8; ++j) a[j]     = (__bf16)p0[j];
            #pragma unroll
            for (int j = 0; j < 8; ++j) a[8 + j] = (__bf16)p0[8 + j];
            afrag[k] = a;
        }
    }
    float zzp[8];
    #pragma unroll
    for (int i = 0; i < 8; ++i) zzp[i] = 0.f;

    #pragma unroll 1
    for (int n = 0; n < 16; ++n) {
        v8f acc = {};
        #pragma unroll
        for (int k = 0; k < 8; ++k) {
            // B (32x16), same contiguous k order as A: lanes 0-15 N=lc K=0..15 (2/VGPR),
            // lanes 16-31 N=lc K=16..31. B[K][N] = Win[n*16+lc][k*32+K].
            const __bf16* bp = winb + (size_t)(n * 16 + lc) * DDIM + k * 32 + half * 16;
            v16bf b = *(const v16bf*)bp;
            acc = wmma_bf16(afrag[k], b, acc);
        }
        const int col = n * 16 + lc;
        #pragma unroll
        for (int i = 0; i < 8; ++i) {          // C: VGPR i -> row half*8+i, col lc
            float v = acc[i];
            hz[w][half * 8 + i][col] = v;
            zzp[i] += v * v;
        }
    }
    // reduce |z|^2 across the 16 column-lanes of each half
    #pragma unroll
    for (int i = 0; i < 8; ++i) {
        float v = zzp[i];
        for (int m = 1; m < 16; m <<= 1) v += __shfl_xor(v, m, 32);
        if (lc == 0) zzs[w][half * 8 + i] = v;
    }

    // ================= Stage B: distances + argmin =====================
    v16bf za[8];
    #pragma unroll
    for (int k = 0; k < 8; ++k) {
        const float* p0 = &hz[w][lc][k * 32 + half * 16];
        v16bf a;
        #pragma unroll
        for (int j = 0; j < 8; ++j) a[j]     = (__bf16)p0[j];
        #pragma unroll
        for (int j = 0; j < 8; ++j) a[8 + j] = (__bf16)p0[8 + j];
        za[k] = a;
    }
    float mind[8]; int mini[8]; float zzv[8];
    #pragma unroll
    for (int i = 0; i < 8; ++i) { mind[i] = 3.4e38f; mini[i] = 0; zzv[i] = zzs[w][half * 8 + i]; }

    #pragma unroll 1
    for (int ct = 0; ct < 32; ++ct) {          // 32 code tiles of 16
        v8f acc = {};
        #pragma unroll
        for (int k = 0; k < 8; ++k) {
            const __bf16* bp = cbb + (size_t)(ct * 16 + lc) * DDIM + k * 32 + half * 16;
            v16bf b = *(const v16bf*)bp;
            acc = wmma_bf16(za[k], b, acc);
        }
        const int code = ct * 16 + lc;
        const float ccv = ccs[code];
        #pragma unroll
        for (int i = 0; i < 8; ++i) {
            float d = zzv[i] - 2.f * acc[i] + ccv;   // ||z-c||^2
            if (d < mind[i]) { mind[i] = d; mini[i] = code; }
        }
    }
    // cross-lane argmin within each 16-lane group (first-index tie-break)
    #pragma unroll
    for (int i = 0; i < 8; ++i) {
        float dmin = mind[i]; int imin = mini[i];
        for (int m = 1; m < 16; m <<= 1) {
            float od = __shfl_xor(dmin, m, 32);
            int   oi = __shfl_xor(imin, m, 32);
            if (od < dmin || (od == dmin && oi < imin)) { dmin = od; imin = oi; }
        }
        if (lc == 0) idxs[w][half * 8 + i] = imin;
    }

    // ================= loss: 1.25 * mean((zq - z)^2) ===================
    float sq = 0.f;
    #pragma unroll 1
    for (int r = 0; r < 16; ++r) {
        const int ci = idxs[w][r];
        const float* crow = codebook + (size_t)ci * DDIM + l * 8;
        const float* zrow = &hz[w][r][l * 8];
        #pragma unroll
        for (int j = 0; j < 8; ++j) { float df = zrow[j] - crow[j]; sq += df * df; }
    }
    for (int m = 1; m < 32; m <<= 1) sq += __shfl_xor(sq, m, 32);
    if (l == 0) atomicAdd(loss, sq * LOSS_SCALE);

    // ================= Stage C: out = zq @ Wout^T, h = hidden + m*out ==
    v16bf qa[8];
    {
        const int ci = idxs[w][lc];
        const __bf16* qrow = cbb + (size_t)ci * DDIM;
        #pragma unroll
        for (int k = 0; k < 8; ++k) {
            const __bf16* p0 = qrow + k * 32 + half * 16;
            v16bf a;
            #pragma unroll
            for (int j = 0; j < 8; ++j) a[j]     = p0[j];
            #pragma unroll
            for (int j = 0; j < 8; ++j) a[8 + j] = p0[8 + j];
            qa[k] = a;
        }
    }
    float fm[8];
    #pragma unroll
    for (int i = 0; i < 8; ++i) fm[i] = amask[rowBase + half * 8 + i] ? 1.f : 0.f;

    #pragma unroll 1
    for (int n = 0; n < 16; ++n) {
        v8f acc = {};
        #pragma unroll
        for (int k = 0; k < 8; ++k) {
            const __bf16* bp = woutb + (size_t)(n * 16 + lc) * DDIM + k * 32 + half * 16;
            v16bf b = *(const v16bf*)bp;
            acc = wmma_bf16(qa[k], b, acc);
        }
        const int col = n * 16 + lc;
        #pragma unroll
        for (int i = 0; i < 8; ++i) {
            const int wrow = half * 8 + i;
            float h = hidden[(size_t)(rowBase + wrow) * DDIM + col] + fm[i] * acc[i];
            hz[w][wrow][col] = h;                 // z no longer needed
        }
    }

    // ================= LayerNorm over D, coalesced store ===============
    #pragma unroll 1
    for (int r = 0; r < 16; ++r) {
        float s = 0.f, s2 = 0.f;
        #pragma unroll
        for (int c = l; c < DDIM; c += 32) { float v = hz[w][r][c]; s += v; s2 += v * v; }
        for (int m = 1; m < 32; m <<= 1) { s += __shfl_xor(s, m, 32); s2 += __shfl_xor(s2, m, 32); }
        const float mu  = s * (1.f / 256.f);
        const float var = s2 * (1.f / 256.f) - mu * mu;
        const float rs  = rsqrtf(var + LN_EPS);
        float* orow = out + (size_t)(rowBase + r) * DDIM;
        const int c0 = l * 8;                    // lane-contiguous 32B stores
        #pragma unroll
        for (int j = 0; j < 8; ++j) {
            const int c = c0 + j;
            orow[c] = (hz[w][r][c] - mu) * rs * gms[c] + bts[c];
        }
    }
}

// ---------------------------------------------------------------------------
extern "C" void kernel_launch(void* const* d_in, const int* in_sizes, int n_in,
                              void* d_out, int out_size, void* d_ws, size_t ws_size,
                              hipStream_t stream) {
    const float*   hidden   = (const float*)d_in[0];
    const uint8_t* amask    = (const uint8_t*)d_in[1];   // bool -> 1 byte
    const float*   codebook = (const float*)d_in[2];
    const float*   W_in     = (const float*)d_in[3];
    const float*   W_out    = (const float*)d_in[4];
    const float*   gamma    = (const float*)d_in[5];
    const float*   beta     = (const float*)d_in[6];

    float* out  = (float*)d_out;
    float* loss = out + (size_t)65536 * 256;             // last element

    // workspace layout (all offsets 256B-aligned)
    char* ws = (char*)d_ws;
    __bf16* winb  = (__bf16*)(ws + 0);                   // 128 KB
    __bf16* woutb = (__bf16*)(ws + 131072);              // 128 KB
    __bf16* cbb   = (__bf16*)(ws + 262144);              // 256 KB
    float*  cc    = (float*) (ws + 524288);              //   2 KB

    vq_init_kernel<<<512, 256, 0, stream>>>(codebook, W_in, W_out,
                                            winb, woutb, cbb, cc, loss);
    vq_fused_kernel<<<NBLOCKS, 256, 0, stream>>>(hidden, amask, codebook,
                                                 gamma, beta, winb, woutb,
                                                 cbb, cc, out, loss);
}
